// DCM_56040733278668
// MI455X (gfx1250) — compile-verified, matrix operand order checked by reading, not run on hardware
//
#include <hip/hip_runtime.h>

#define B_    16
#define INC   256
#define MIDC  64
#define OUTC  256
#define H_    96
#define W_    96
#define HW    9216      // 96*96
#define CATC  448       // 256 + 3*64

typedef __attribute__((ext_vector_type(16))) _Float16 v16h;
typedef __attribute__((ext_vector_type(8)))  float    v8f;

union FragH { v16h v; unsigned u[8]; };

__device__ __forceinline__ v8f wmma_f16(const FragH& a, const FragH& b, v8f c) {
  // D = A(16x32 f16) * B(32x16 f16) + C(16x16 f32)
  return __builtin_amdgcn_wmma_f32_16x16x32_f16(false, a.v, false, b.v,
                                                (short)0, c, false, false);
}

// K-pair pattern for 16-bit 16x32 A fragment (mirrored for B):
// VGPR jj in 0..3 -> K = 2*jj,2*jj+1 ; jj in 4..7 -> K = 16+2*(jj-4)..
// lanes 16..31 add +8 to K.
__device__ __forceinline__ int kofs(int jj, int lane) {
  return ((jj & 3) << 1) + ((jj >> 2) << 4) + ((lane >> 4) << 3);
}

// -------- async global->LDS (GLOBAL_LOAD_ASYNC_TO_LDS_B128, ASYNCcnt) --------
#if __has_builtin(__builtin_amdgcn_global_load_async_to_lds_b128) && \
    __has_builtin(__builtin_amdgcn_s_wait_asynccnt)
#define USE_ASYNC 1
typedef int v4i_ __attribute__((vector_size(16)));
typedef __attribute__((address_space(1))) v4i_ as1_v4i;   // global
typedef __attribute__((address_space(3))) v4i_ as3_v4i;   // LDS
// copies 8 f16 (16 bytes); both addresses must be 16B aligned
__device__ __forceinline__ void async_cp16B(const _Float16* g, const _Float16* l) {
  as1_v4i* gp = (as1_v4i*)(uintptr_t)g;              // AS1 value == generic VA
  as3_v4i* lp = (as3_v4i*)(unsigned)(uintptr_t)l;    // AS3 value == VA[31:0]
  __builtin_amdgcn_global_load_async_to_lds_b128(gp, lp, 0, 0);
}
__device__ __forceinline__ void async_wait0() {
  __builtin_amdgcn_s_wait_asynccnt(0);
}
#else
#define USE_ASYNC 0
#endif

// ---------------------------------------------------------------- weights f32 -> f16
__global__ void cvt_f16_kernel(const float* __restrict__ src,
                               _Float16* __restrict__ dst, int n) {
  int i = blockIdx.x * blockDim.x + threadIdx.x;
  if (i < n) dst[i] = (_Float16)src[i];
}

// ---------------------------------------------------------------- adaptive avg pool 3x3
// one wave per (b, c, tap) bin; each bin is 32x32 contiguous
__global__ __launch_bounds__(256) void pool_kernel(const float* __restrict__ y,
                                                   float* __restrict__ pooled) {
  int wid  = (blockIdx.x * 256 + threadIdx.x) >> 5;
  int lane = threadIdx.x & 31;
  if (wid >= B_ * INC * 9) return;
  int tap = wid % 9;
  int bc  = wid / 9;
  int ky = tap / 3, kx = tap % 3;
  const float* row = y + (size_t)bc * HW + (size_t)(ky * 32 + lane) * W_ + kx * 32;
  float s = 0.f;
#pragma unroll
  for (int j = 0; j < 32; ++j) s += row[j];
#pragma unroll
  for (int off = 16; off >= 1; off >>= 1) s += __shfl_xor(s, off, 32);
  if (lane == 0) pooled[(size_t)bc * 9 + tap] = s * (1.0f / 1024.0f);
}

// ---------------------------------------------------------------- dynamic 3x3 kernels
__global__ void dynk_kernel(const float* __restrict__ pooled,
                            const float* __restrict__ w_gk,
                            const float* __restrict__ b_gk,
                            float* __restrict__ dynk) {
  int gid = blockIdx.x * blockDim.x + threadIdx.x;
  if (gid >= B_ * MIDC * 9) return;
  int tap = gid % 9;
  int m   = (gid / 9) % MIDC;
  int b   = gid / (9 * MIDC);
  float acc = b_gk[m];
  const float* wr = w_gk + (size_t)m * INC;
  const float* pr = pooled + (size_t)b * INC * 9 + tap;
  for (int c = 0; c < INC; ++c) acc += wr[c] * pr[c * 9];
  dynk[gid] = acc;
}

// ---------------------------------------------------------------- trans: xin = Wtr(64x256) * x + b
// block tile 64(M) x 256(N); 8 waves = 4(M) x 2(N-half); wave: 1 m-frag x 8 n-frags
__global__ __launch_bounds__(256) void trans_gemm_kernel(
    const float* __restrict__ x, const _Float16* __restrict__ wtr,
    const float* __restrict__ b_tr, float* __restrict__ xin) {
  __shared__ _Float16 sA[MIDC][40];   // weights [m][k], 16B-aligned rows
  __shared__ _Float16 sB[256][40];    // activations [n][k]
  const int t = threadIdx.x, lane = t & 31, wave = t >> 5;
  const int b   = blockIdx.x / 36;
  const int hw0 = (blockIdx.x % 36) * 256;
  const int wm = wave & 3, wn = wave >> 2;

  v8f acc[8];
#pragma unroll
  for (int j = 0; j < 8; ++j) acc[j] = v8f{};

  for (int kt = 0; kt < INC / 32; ++kt) {
    const int kb = kt * 32;
#if USE_ASYNC
    {                                         // A: 64x32 f16, 256 x 16B transfers
      int m = t >> 2, q = t & 3;
      async_cp16B(&wtr[(size_t)m * INC + kb + q * 8], &sA[m][q * 8]);
    }
#else
#pragma unroll
    for (int i = 0; i < 8; ++i) {
      int idx = t + i * 256;
      sA[idx >> 5][idx & 31] = wtr[(size_t)(idx >> 5) * INC + kb + (idx & 31)];
    }
#endif
#pragma unroll
    for (int i = 0; i < 32; ++i) {            // B: 32 channels x 256 cols (coalesced f32)
      sB[t][i] = (_Float16)x[((size_t)(b * INC + kb + i)) * HW + hw0 + t];
    }
#if USE_ASYNC
    async_wait0();
#endif
    __syncthreads();

    FragH af;
#pragma unroll
    for (int jj = 0; jj < 8; ++jj)
      af.u[jj] = *(const unsigned*)&sA[wm * 16 + (lane & 15)][kofs(jj, lane)];
#pragma unroll
    for (int j = 0; j < 8; ++j) {
      FragH bf;
      int n = wn * 128 + j * 16 + (lane & 15);
#pragma unroll
      for (int jj = 0; jj < 8; ++jj)
        bf.u[jj] = *(const unsigned*)&sB[n][kofs(jj, lane)];
      acc[j] = wmma_f16(af, bf, acc[j]);
    }
    __syncthreads();
  }

  const int hi = lane >> 4, nc0 = lane & 15;
#pragma unroll
  for (int j = 0; j < 8; ++j) {
    int n = hw0 + wn * 128 + j * 16 + nc0;
#pragma unroll
    for (int r = 0; r < 8; ++r) {
      int oc = wm * 16 + hi * 8 + r;
      xin[((size_t)(b * MIDC + oc)) * HW + n] = acc[j][r] + b_tr[oc];
    }
  }
}

// ---------------------------------------------------------------- fused dilated depthwise 3x3
// + fuse_inside 64x64 GEMM.  block: 64 channels x 128 pixels of one image.
__global__ __launch_bounds__(256) void dwfuse_kernel(
    const float* __restrict__ xin, const float* __restrict__ dynk,
    const _Float16* __restrict__ wfi, const float* __restrict__ b_fi,
    _Float16* __restrict__ fout, int dil) {
  __shared__ float    sK[MIDC][9];
  __shared__ _Float16 sA[MIDC][72];   // w_fi [m][k], full 64x64, 16B-aligned rows
  __shared__ _Float16 sB[128][72];    // depthwise result [pix][c]
  const int t = threadIdx.x, lane = t & 31, wave = t >> 5;
  const int b   = blockIdx.x / 72;
  const int hw0 = (blockIdx.x % 72) * 128;

#if USE_ASYNC
#pragma unroll
  for (int i = 0; i < 2; ++i) {               // 64x64 f16 = 512 x 16B transfers
    int idx = t + i * 256;
    int m = idx >> 3, q = idx & 7;
    async_cp16B(&wfi[(size_t)m * MIDC + q * 8], &sA[m][q * 8]);
  }
#else
#pragma unroll
  for (int i = 0; i < 16; ++i) {
    int idx = t + i * 256;
    sA[idx >> 6][idx & 63] = wfi[idx];
  }
#endif
  for (int idx = t; idx < MIDC * 9; idx += 256)
    sK[idx / 9][idx % 9] = dynk[(size_t)b * MIDC * 9 + idx];
#if USE_ASYNC
  async_wait0();
#endif
  __syncthreads();

  // depthwise: each thread 32 (c,pix) outputs, pixels contiguous per iteration
#pragma unroll
  for (int i = 0; i < 32; ++i) {
    int idx = t + i * 256;
    int c = idx >> 7, pix = idx & 127;
    int hw = hw0 + pix, h = hw / 96, w = hw % 96;
    const float* src = xin + ((size_t)(b * MIDC + c)) * HW;
    float a = 0.f;
#pragma unroll
    for (int ky = 0; ky < 3; ++ky) {
      int hh = h + (ky - 1) * dil;
      if ((unsigned)hh >= 96u) continue;
#pragma unroll
      for (int kx = 0; kx < 3; ++kx) {
        int ww = w + (kx - 1) * dil;
        if ((unsigned)ww >= 96u) continue;
        a += sK[c][ky * 3 + kx] * src[hh * 96 + ww];
      }
    }
    sB[pix][c] = (_Float16)a;
  }
  __syncthreads();

  // fuse_inside GEMM: 64(M) x 128(N), K=64 in 2 steps
  const int wm = wave & 3, wn = wave >> 2;
  v8f acc[4];
#pragma unroll
  for (int j = 0; j < 4; ++j) acc[j] = v8f{};
#pragma unroll
  for (int kt = 0; kt < 2; ++kt) {
    int kb = kt * 32;
    FragH af;
#pragma unroll
    for (int jj = 0; jj < 8; ++jj)
      af.u[jj] = *(const unsigned*)&sA[wm * 16 + (lane & 15)][kb + kofs(jj, lane)];
#pragma unroll
    for (int j = 0; j < 4; ++j) {
      FragH bf;
      int n = wn * 64 + j * 16 + (lane & 15);
#pragma unroll
      for (int jj = 0; jj < 8; ++jj)
        bf.u[jj] = *(const unsigned*)&sB[n][kb + kofs(jj, lane)];
      acc[j] = wmma_f16(af, bf, acc[j]);
    }
  }

  const int hi = lane >> 4, nc0 = lane & 15;
#pragma unroll
  for (int j = 0; j < 4; ++j) {
    int n = hw0 + wn * 64 + j * 16 + nc0;
#pragma unroll
    for (int r = 0; r < 8; ++r) {
      int oc = wm * 16 + hi * 8 + r;
      fout[((size_t)(b * MIDC + oc)) * HW + n] = (_Float16)(acc[j][r] + b_fi[oc]);
    }
  }
}

// ---------------------------------------------------------------- fuse_outside:
// out = Wfo(256x448) * cat[x; f0; f1; f2] + b.  block tile 256(M) x 128(N),
// wave: 2 m-frags x 8 n-frags = 16 WMMAs per K-step, 14 K-steps.
__global__ __launch_bounds__(256) void final_gemm_kernel(
    const float* __restrict__ x,
    const _Float16* __restrict__ f0, const _Float16* __restrict__ f1,
    const _Float16* __restrict__ f2,
    const _Float16* __restrict__ wfo, const float* __restrict__ b_fo,
    float* __restrict__ out) {
  __shared__ _Float16 sA[OUTC][40];
  __shared__ _Float16 sB[128][40];
  const int t = threadIdx.x, lane = t & 31, wave = t >> 5;
  const int b   = blockIdx.x / 72;
  const int hw0 = (blockIdx.x % 72) * 128;

  v8f acc[2][8];
#pragma unroll
  for (int r2 = 0; r2 < 2; ++r2)
#pragma unroll
    for (int j = 0; j < 8; ++j) acc[r2][j] = v8f{};

  for (int kt = 0; kt < CATC / 32; ++kt) {
    const int kb = kt * 32;
#if USE_ASYNC
#pragma unroll
    for (int i = 0; i < 4; ++i) {             // A: 256x32 f16 = 1024 x 16B transfers
      int idx = t + i * 256;
      int m = idx >> 2, q = idx & 3;
      async_cp16B(&wfo[(size_t)m * CATC + kb + q * 8], &sA[m][q * 8]);
    }
#else
#pragma unroll
    for (int i = 0; i < 32; ++i) {
      int idx = t + i * 256;
      sA[idx >> 5][idx & 31] = wfo[(size_t)(idx >> 5) * CATC + kb + (idx & 31)];
    }
#endif
#pragma unroll
    for (int i = 0; i < 16; ++i) {            // B: 32 rows x 128 cols, piecewise source
      int kk = (i << 1) | (t >> 7);
      int n  = t & 127;
      int k  = kb + kk;
      _Float16 v;
      if (k < INC) {
        v = (_Float16)x[((size_t)(b * INC + k)) * HW + hw0 + n];
      } else {
        int kc = k - INC;
        const _Float16* fp = (kc < 64) ? f0 : ((kc < 128) ? f1 : f2);
        v = fp[((size_t)(b * MIDC + (kc & 63))) * HW + hw0 + n];
      }
      sB[n][kk] = v;
    }
#if USE_ASYNC
    async_wait0();
#endif
    __syncthreads();

    FragH af[2];
#pragma unroll
    for (int r2 = 0; r2 < 2; ++r2)
#pragma unroll
      for (int jj = 0; jj < 8; ++jj)
        af[r2].u[jj] =
            *(const unsigned*)&sA[wave * 32 + r2 * 16 + (lane & 15)][kofs(jj, lane)];
#pragma unroll
    for (int j = 0; j < 8; ++j) {
      FragH bf;
      int n = j * 16 + (lane & 15);
#pragma unroll
      for (int jj = 0; jj < 8; ++jj)
        bf.u[jj] = *(const unsigned*)&sB[n][kofs(jj, lane)];
      acc[0][j] = wmma_f16(af[0], bf, acc[0][j]);
      acc[1][j] = wmma_f16(af[1], bf, acc[1][j]);
    }
    __syncthreads();
  }

  const int hi = lane >> 4, nc0 = lane & 15;
#pragma unroll
  for (int r2 = 0; r2 < 2; ++r2)
#pragma unroll
    for (int j = 0; j < 8; ++j) {
      int n = hw0 + j * 16 + nc0;
#pragma unroll
      for (int r = 0; r < 8; ++r) {
        int oc = wave * 32 + r2 * 16 + hi * 8 + r;
        out[((size_t)(b * OUTC + oc)) * HW + n] = acc[r2][j][r] + b_fo[oc];
      }
    }
}

// ----------------------------------------------------------------
extern "C" void kernel_launch(void* const* d_in, const int* in_sizes, int n_in,
                              void* d_out, int out_size, void* d_ws, size_t ws_size,
                              hipStream_t stream) {
  (void)in_sizes; (void)n_in; (void)out_size; (void)ws_size;
  const float* x    = (const float*)d_in[0];
  const float* y    = (const float*)d_in[1];
  const float* w_gk = (const float*)d_in[2];
  const float* b_gk = (const float*)d_in[3];
  const float* w_tr = (const float*)d_in[4];
  const float* b_tr = (const float*)d_in[5];
  const float* w_fi = (const float*)d_in[6];
  const float* b_fi = (const float*)d_in[7];
  const float* w_fo = (const float*)d_in[8];
  const float* b_fo = (const float*)d_in[9];
  float* out = (float*)d_out;

  char* ws = (char*)d_ws;
  size_t off = 0;
  auto alloc = [&](size_t bytes) -> void* {
    void* p = ws + off;
    off += (bytes + 255) & ~(size_t)255;
    return p;
  };
  _Float16* wtr_h = (_Float16*)alloc((size_t)MIDC * INC * 2);
  _Float16* wfi_h = (_Float16*)alloc((size_t)MIDC * MIDC * 2);
  _Float16* wfo_h = (_Float16*)alloc((size_t)OUTC * CATC * 2);
  float* pooled   = (float*)alloc((size_t)B_ * INC * 9 * 4);
  float* dynk     = (float*)alloc((size_t)B_ * MIDC * 9 * 4);
  float* xin      = (float*)alloc((size_t)B_ * MIDC * HW * 4);
  _Float16* fb0   = (_Float16*)alloc((size_t)B_ * MIDC * HW * 2);
  _Float16* fb1   = (_Float16*)alloc((size_t)B_ * MIDC * HW * 2);
  _Float16* fb2   = (_Float16*)alloc((size_t)B_ * MIDC * HW * 2);

  cvt_f16_kernel<<<(MIDC * INC + 255) / 256, 256, 0, stream>>>(w_tr, wtr_h, MIDC * INC);
  cvt_f16_kernel<<<(MIDC * MIDC + 255) / 256, 256, 0, stream>>>(w_fi, wfi_h, MIDC * MIDC);
  cvt_f16_kernel<<<(OUTC * CATC + 255) / 256, 256, 0, stream>>>(w_fo, wfo_h, OUTC * CATC);

  pool_kernel<<<(B_ * INC * 9 * 32) / 256, 256, 0, stream>>>(y, pooled);
  dynk_kernel<<<(B_ * MIDC * 9 + 255) / 256, 256, 0, stream>>>(pooled, w_gk, b_gk, dynk);

  trans_gemm_kernel<<<B_ * (HW / 256), 256, 0, stream>>>(x, wtr_h, b_tr, xin);

  dwfuse_kernel<<<B_ * (HW / 128), 256, 0, stream>>>(xin, dynk, wfi_h, b_fi, fb0, 1);
  dwfuse_kernel<<<B_ * (HW / 128), 256, 0, stream>>>(xin, dynk, wfi_h, b_fi, fb1, 3);
  dwfuse_kernel<<<B_ * (HW / 128), 256, 0, stream>>>(xin, dynk, wfi_h, b_fi, fb2, 5);

  final_gemm_kernel<<<B_ * (HW / 128), 256, 0, stream>>>(x, fb0, fb1, fb2, wfo_h, b_fo, out);
}
